// Voxel2Mesh_25202868093094
// MI455X (gfx1250) — matrix operand order repaired and checked by reading, compiled, not run on hardware
//
#include <hip/hip_runtime.h>
#include <math.h>

// ---------------------------------------------------------------------------
// Types for CDNA5 WMMA (wave32, 16x16x32 bf16 -> f32)
// ---------------------------------------------------------------------------
typedef __attribute__((ext_vector_type(16))) __bf16 v16bf;
typedef __attribute__((ext_vector_type(8)))  float  v8f;
typedef __attribute__((ext_vector_type(8)))  int    v8i;

#define DEVINL __device__ __forceinline__

DEVINL __bf16 f2bf(float f) {
  unsigned u = __builtin_bit_cast(unsigned, f);
  unsigned r = u + 0x7FFFu + ((u >> 16) & 1u);   // round-to-nearest-even
  unsigned short h = (unsigned short)(r >> 16);
  return __builtin_bit_cast(__bf16, h);
}

// Pack two f32 -> two bf16 in one dword: round-half-up + v_perm_b32
DEVINL int pack2bf(float f0, float f1) {
  unsigned u0 = __builtin_bit_cast(unsigned, f0) + 0x8000u;
  unsigned u1 = __builtin_bit_cast(unsigned, f1) + 0x8000u;
  return (int)__builtin_amdgcn_perm(u1, u0, 0x07060302u);  // {u1.hi16, u0.hi16}
}

// ---------------------------------------------------------------------------
// Pack fp32 conv weights (Co, Ci, 27) into bf16 WMMA A fragments with
// TAP-MAJOR K ordering: logical k = tap*Ci + ci  (Ci is a power of two).
// A-fragment 16x32 bf16 layout as packed here (contiguous k): lane L: m = L&15, half = L>>4;
// packed element e holds K index k = e + 16*half.
// apack[((ct*nkc + kc)*32 + lane)*16 + e]
// ---------------------------------------------------------------------------
__global__ void pack_wmma_a(const float* __restrict__ w, __bf16* __restrict__ apack,
                            int Co, int Ci, int ciShift) {
  int K = Ci * 27;
  int nkc = (K + 31) >> 5;
  int nct = (Co + 15) >> 4;
  long total = (long)nct * nkc * 512;
  long idx = (long)blockIdx.x * blockDim.x + threadIdx.x;
  if (idx >= total) return;
  int e    = (int)(idx & 15);
  int lane = (int)((idx >> 4) & 31);
  long rest = idx >> 9;
  int kc = (int)(rest % nkc);
  int ct = (int)(rest / nkc);
  int m = lane & 15, half = lane >> 4;
  int k  = (kc << 5) + e + (half << 4);
  int co = (ct << 4) + m;
  float v = 0.0f;
  if (co < Co && k < K) {
    int ci = k & (Ci - 1);
    int t  = k >> ciShift;          // tap index 0..26
    v = w[(long)co * K + (long)ci * 27 + t];
  }
  apack[idx] = f2bf(v);
}

// ---------------------------------------------------------------------------
// f32 NCDHW volume -> bf16 pair-interleaved SPATIAL-MAJOR volume:
//   pvol[p*Cp + q] = { bf16(in[(2q+1)*DHW+p]), bf16(in[2q*DHW+p]) }
// Cp = Ci/2 (multiple of 8) => any aligned 8-dword run [q0, q0+8) within a
// row is 32-byte aligned and contiguous -> one v8i load per B fragment.
// ---------------------------------------------------------------------------
__global__ void pack_pairs(const float* __restrict__ in, unsigned* __restrict__ out,
                           int Cp, long DHW) {
  long total = (long)Cp * DHW;
  long i = (long)blockIdx.x * blockDim.x + threadIdx.x;
  if (i >= total) return;
  long q = i % Cp;
  long p = i / Cp;
  out[i] = (unsigned)pack2bf(in[(2 * q) * DHW + p], in[(2 * q + 1) * DHW + p]);
}

// ---------------------------------------------------------------------------
// Implicit-GEMM 3x3x3 conv, pad 1, stride 1, cubic dim d (d^3 % 16 == 0).
// Input: bf16 pair-interleaved spatial-major volume. One wave per block:
// NT x 16 output channels x 16 spatial outputs. K = Ci*27 tap-major: each
// 16-wide K half-chunk has a single tap per lane, so the whole B fragment is
// ONE 32-byte vector load. B double-buffered across chunks; NT A fragments
// loaded as a clause before NT back-to-back WMMAs. Epilogue is branch-free.
// B-fragment 32x16 bf16: N = lane&15, element e holds K = e + 16*(lane>>4).
// D layout: lane L, acc element r -> M = r + 8*(L>>4), N = L&15.
// ---------------------------------------------------------------------------
template <int NT>
__global__ __launch_bounds__(32)
void conv3_wmma_pair(const unsigned* __restrict__ pin, const __bf16* __restrict__ apack,
                     const float* __restrict__ bias, float* __restrict__ out,
                     int Ci, int ciShift, int d, int relu) {
  const int DHW = d * d * d;
  const int K   = Ci * 27;
  const int nkc = (K + 31) >> 5;
  const int Cp  = Ci >> 1;
  const int lane  = threadIdx.x;
  const int nBase = blockIdx.x << 4;
  const int tile0 = blockIdx.y * NT;

  const int col  = lane & 15;
  const int half = lane >> 4;
  const int p  = nBase + col;
  const int px = p % d;
  const int py = (p / d) % d;
  const int pz = p / (d * d);

  auto loadB = [&](int kc) -> v8i {
    int kbase = (kc << 5) + (half << 4);
    bool kvalid = (kbase < K);
    int t  = kbase >> ciShift;
    int q0 = (kbase & (Ci - 1)) >> 1;
    int dz = t / 9, rem = t - dz * 9;
    int dy = rem / 3, dx = rem - dy * 3;
    int z = pz + dz - 1, y = py + dy - 1, x = px + dx - 1;
    bool v = kvalid &&
             (unsigned)z < (unsigned)d && (unsigned)y < (unsigned)d &&
             (unsigned)x < (unsigned)d;
    v8i r = {};
    if (v)
      r = *(const v8i*)(pin + ((long)((z * d + y) * d + x) * Cp + q0));
    return r;
  };

  v8f acc[NT];
#pragma unroll
  for (int c = 0; c < NT; ++c) acc[c] = (v8f){};

  const __bf16* apBase = apack + (long)lane * 16;

  v8i bcur = loadB(0);
  for (int kc = 0; kc < nkc; ++kc) {
    v8i bnext = {};
    if (kc + 1 < nkc) bnext = loadB(kc + 1);
    v16bf b = __builtin_bit_cast(v16bf, bcur);
    v16bf a[NT];
#pragma unroll
    for (int c = 0; c < NT; ++c)
      a[c] = *(const v16bf*)(apBase + ((long)(tile0 + c) * nkc + kc) * 512);
#pragma unroll
    for (int c = 0; c < NT; ++c)
      acc[c] = __builtin_amdgcn_wmma_f32_16x16x32_bf16(false, a[c], false, b,
                                                       (short)0, acc[c],
                                                       false, false);
    bcur = bnext;
  }

  // Branch-free epilogue: d^3 divisible by 16 and exact grids make all
  // bounds checks provably true.
  const int coBase = tile0 * 16 + (half << 3);
  float bv[NT * 8];
#pragma unroll
  for (int c = 0; c < NT; ++c)
#pragma unroll
    for (int r = 0; r < 8; ++r) bv[c * 8 + r] = bias[coBase + c * 16 + r];
#pragma unroll
  for (int c = 0; c < NT; ++c) {
#pragma unroll
    for (int r = 0; r < 8; ++r) {
      float vv = acc[c][r] + bv[c * 8 + r];
      if (relu) vv = fmaxf(vv, 0.0f);
      out[(long)(coBase + c * 16 + r) * DHW + p] = vv;
    }
  }
}

// ---------------------------------------------------------------------------
// First conv only: Ci == 1, K = 27 (one chunk), f32 input, Co = 16.
// ---------------------------------------------------------------------------
__global__ __launch_bounds__(32)
void conv3_wmma_c1(const float* __restrict__ in, const __bf16* __restrict__ apack,
                   const float* __restrict__ bias, float* __restrict__ out,
                   int d, int relu) {
  const int DHW = d * d * d;
  const int lane  = threadIdx.x;
  const int nBase = blockIdx.x << 4;
  const int col  = lane & 15;
  const int half = lane >> 4;
  const int p  = nBase + col;
  const int px = p % d;
  const int py = (p / d) % d;
  const int pz = p / (d * d);

  v8i bi;
#pragma unroll
  for (int j = 0; j < 8; ++j) {
    float f[2];
#pragma unroll
    for (int s = 0; s < 2; ++s) {
      int k = 2 * j + s + (half << 4);
      float vv = 0.0f;
      if (k < 27) {
        int dz = k / 9, rem = k - dz * 9;
        int dy = rem / 3, dx = rem - dy * 3;
        int z = pz + dz - 1, y = py + dy - 1, x = px + dx - 1;
        if ((unsigned)z < (unsigned)d && (unsigned)y < (unsigned)d &&
            (unsigned)x < (unsigned)d)
          vv = in[((long)z * d + y) * d + x];
      }
      f[s] = vv;
    }
    bi[j] = pack2bf(f[0], f[1]);
  }
  v16bf b = __builtin_bit_cast(v16bf, bi);
  v16bf a = *(const v16bf*)(apack + (long)lane * 16);
  v8f acc = {};
  acc = __builtin_amdgcn_wmma_f32_16x16x32_bf16(false, a, false, b, (short)0,
                                                acc, false, false);
  const int coBase = half << 3;
#pragma unroll
  for (int r = 0; r < 8; ++r) {
    float vv = acc[r] + bias[coBase + r];
    if (relu) vv = fmaxf(vv, 0.0f);
    out[(long)(coBase + r) * DHW + p] = vv;
  }
}

// ---------------------------------------------------------------------------
// 2x2x2 max pool, stride 2
// ---------------------------------------------------------------------------
__global__ void maxpool2(const float* __restrict__ in, float* __restrict__ out,
                         int C, int d) {
  int od = d >> 1;
  long total = (long)C * od * od * od;
  long i = (long)blockIdx.x * blockDim.x + threadIdx.x;
  if (i >= total) return;
  int x = (int)(i % od);
  int y = (int)((i / od) % od);
  int z = (int)((i / ((long)od * od)) % od);
  int c = (int)(i / ((long)od * od * od));
  const float* pbase = in + (((long)c * d + 2 * z) * d + 2 * y) * d + 2 * x;
  float m = -INFINITY;
#pragma unroll
  for (int dz = 0; dz < 2; ++dz)
#pragma unroll
    for (int dy = 0; dy < 2; ++dy)
#pragma unroll
      for (int dx = 0; dx < 2; ++dx)
        m = fmaxf(m, pbase[((long)dz * d + dy) * d + dx]);
  out[i] = m;
}

// ---------------------------------------------------------------------------
// conv_transpose 2x2x2 stride 2, weight (Ci,Co,2,2,2) 'IODHW'
// ---------------------------------------------------------------------------
__global__ void upconv2(const float* __restrict__ in, const float* __restrict__ w,
                        const float* __restrict__ b, float* __restrict__ out,
                        int Ci, int Co, int di) {
  int dop = di << 1;
  long total = (long)Co * dop * dop * dop;
  long i = (long)blockIdx.x * blockDim.x + threadIdx.x;
  if (i >= total) return;
  int xo = (int)(i % dop);
  int yo = (int)((i / dop) % dop);
  int zo = (int)((i / ((long)dop * dop)) % dop);
  int co = (int)(i / ((long)dop * dop * dop));
  int xi = xo >> 1, yi = yo >> 1, zi = zo >> 1;
  int off = ((zo & 1) << 2) | ((yo & 1) << 1) | (xo & 1);
  float acc = b[co];
  long inoff = ((long)zi * di + yi) * di + xi;
  long istride = (long)di * di * di;
  for (int ci = 0; ci < Ci; ++ci)
    acc += in[(long)ci * istride + inoff] * w[((long)ci * Co + co) * 8 + off];
  out[i] = acc;
}

// ---------------------------------------------------------------------------
// 1x1x1 final conv
// ---------------------------------------------------------------------------
__global__ void conv1x1(const float* __restrict__ h, const float* __restrict__ w,
                        const float* __restrict__ b, float* __restrict__ out,
                        int Ci, int Co, long DHW) {
  long total = (long)Co * DHW;
  long i = (long)blockIdx.x * blockDim.x + threadIdx.x;
  if (i >= total) return;
  long p = i % DHW;
  int nc = (int)(i / DHW);
  float acc = b[nc];
  for (int ci = 0; ci < Ci; ++ci) acc += h[(long)ci * DHW + p] * w[nc * Ci + ci];
  out[i] = acc;
}

// ---------------------------------------------------------------------------
// Adjacency build (162x162)
// ---------------------------------------------------------------------------
__global__ void adj_zero(float* A, int n) {
  long i = (long)blockIdx.x * blockDim.x + threadIdx.x;
  if (i < (long)n * n) A[i] = 0.0f;
}
__global__ void adj_edges(const int* __restrict__ faces, float* A, int nf, int n) {
  int i = blockIdx.x * blockDim.x + threadIdx.x;
  if (i >= nf * 3) return;
  int f = i / 3, e = i % 3;
  int a = faces[f * 3 + e];
  int b = faces[f * 3 + ((e + 1) % 3)];
  A[(long)a * n + b] = 1.0f;
  A[(long)b * n + a] = 1.0f;
}
__global__ void adj_norm(float* A, int n) {
  int r = blockIdx.x * blockDim.x + threadIdx.x;
  if (r >= n) return;
  A[(long)r * n + r] += 1.0f;
  float s = 0.0f;
  for (int j = 0; j < n; ++j) s += A[(long)r * n + j];
  float inv = 1.0f / s;
  for (int j = 0; j < n; ++j) A[(long)r * n + j] *= inv;
}

// ---------------------------------------------------------------------------
// Trilinear sampling: vol [C,d,d,d], pts [M,3] in [-1,1] -> out [M,C]
// ---------------------------------------------------------------------------
__global__ void trilinear(const float* __restrict__ vol, int C, int d,
                          const float* __restrict__ pts, int M,
                          float* __restrict__ out) {
  long total = (long)M * C;
  long i = (long)blockIdx.x * blockDim.x + threadIdx.x;
  if (i >= total) return;
  int c = (int)(i % C);
  int m = (int)(i / C);
  float scale = 0.5f * (float)(d - 1);
  float x = fminf(fmaxf((pts[3 * m + 0] + 1.0f) * scale, 0.0f), (float)(d - 1));
  float y = fminf(fmaxf((pts[3 * m + 1] + 1.0f) * scale, 0.0f), (float)(d - 1));
  float z = fminf(fmaxf((pts[3 * m + 2] + 1.0f) * scale, 0.0f), (float)(d - 1));
  int x0 = min(max((int)floorf(x), 0), d - 2), x1 = x0 + 1;
  int y0 = min(max((int)floorf(y), 0), d - 2), y1 = y0 + 1;
  int z0 = min(max((int)floorf(z), 0), d - 2), z1 = z0 + 1;
  float wx = x - (float)x0, wy = y - (float)y0, wz = z - (float)z0;
  const float* v = vol + (long)c * d * d * d;
  long dd = (long)d * d;
#define G(zi, yi, xi) v[(long)(zi) * dd + (long)(yi) * d + (xi)]
  float r =
      G(z0, y0, x0) * (1 - wz) * (1 - wy) * (1 - wx) + G(z0, y0, x1) * (1 - wz) * (1 - wy) * wx +
      G(z0, y1, x0) * (1 - wz) * wy * (1 - wx)       + G(z0, y1, x1) * (1 - wz) * wy * wx +
      G(z1, y0, x0) * wz * (1 - wy) * (1 - wx)       + G(z1, y0, x1) * wz * (1 - wy) * wx +
      G(z1, y1, x0) * wz * wy * (1 - wx)             + G(z1, y1, x1) * wz * wy * wx;
#undef G
  out[(long)m * C + c] = r;
}

// ---------------------------------------------------------------------------
// Learnt shift deltas + neighbourhood points
// ---------------------------------------------------------------------------
__global__ void build_neigh(const float* __restrict__ center, const float* __restrict__ sd_w,
                            const float* __restrict__ sd_b, const float* __restrict__ verts,
                            float* __restrict__ neigh, int M, int C, int d) {
  int i = blockIdx.x * blockDim.x + threadIdx.x;
  if (i >= M * 27) return;
  int t = i % 27, m = i / 27;
  float s0 = sd_b[t * 3 + 0], s1 = sd_b[t * 3 + 1], s2 = sd_b[t * 3 + 2];
  for (int c = 0; c < C; ++c) {
    float cv = center[(long)m * C + c];
    s0 += cv * sd_w[(long)c * 81 + t * 3 + 0];
    s1 += cv * sd_w[(long)c * 81 + t * 3 + 1];
    s2 += cv * sd_w[(long)c * 81 + t * 3 + 2];
  }
  if (t == 13) { s0 = 0.0f; s1 = 0.0f; s2 = 0.0f; }
  int a = t / 9, b = (t / 3) % 3, c3 = t % 3;
  float step = 2.0f / (float)(d - 1);
  neigh[(long)i * 3 + 0] = verts[m * 3 + 0] + (float)(a - 1) * step + s0;
  neigh[(long)i * 3 + 1] = verts[m * 3 + 1] + (float)(b - 1) * step + s1;
  neigh[(long)i * 3 + 2] = verts[m * 3 + 2] + (float)(c3 - 1) * step + s2;
}

// ---------------------------------------------------------------------------
// Neighbourhood 1x27 conv
// ---------------------------------------------------------------------------
__global__ void neigh_sum(const float* __restrict__ feats, const float* __restrict__ sum_w,
                          const float* __restrict__ sum_b, float* __restrict__ out,
                          int M, int C) {
  long total = (long)M * C;
  long i = (long)blockIdx.x * blockDim.x + threadIdx.x;
  if (i >= total) return;
  int o = (int)(i % C);
  int n = (int)(i / C);
  float acc = sum_b[o];
  for (int c = 0; c < C; ++c) {
    const float* fw = sum_w + ((long)o * C + c) * 27;
    const float* fv = feats + (long)n * 27 * C + c;
    for (int k = 0; k < 27; ++k) acc += fv[(long)k * C] * fw[k];
  }
  out[(long)n * C + o] = acc;
}

// ---------------------------------------------------------------------------
// Tiny generic GEMM: out[M,N] = (acc ? out : bias) + X[M,K] @ W[K,N], opt ReLU
// ---------------------------------------------------------------------------
__global__ void gemm_small(const float* __restrict__ X, const float* __restrict__ W,
                           const float* __restrict__ bias, float* __restrict__ out,
                           int M, int K, int N, int accumulate, int relu) {
  long total = (long)M * N;
  long i = (long)blockIdx.x * blockDim.x + threadIdx.x;
  if (i >= total) return;
  int n = (int)(i % N);
  int m = (int)(i / N);
  float acc = accumulate ? out[i] : (bias ? bias[n] : 0.0f);
  for (int k = 0; k < K; ++k) acc += X[(long)m * K + k] * W[(long)k * N + n];
  if (relu) acc = fmaxf(acc, 0.0f);
  out[i] = acc;
}

__global__ void concat_cols(const float* s0, int w0, const float* s1, int w1,
                            const float* s2, int w2, float* __restrict__ out, int M) {
  int ctot = w0 + w1 + w2;
  long total = (long)M * ctot;
  long i = (long)blockIdx.x * blockDim.x + threadIdx.x;
  if (i >= total) return;
  int c = (int)(i % ctot);
  int m = (int)(i / ctot);
  float v;
  if (c < w0)            v = s0[(long)m * w0 + c];
  else if (c < w0 + w1)  v = s1[(long)m * w1 + (c - w0)];
  else                   v = s2[(long)m * w2 + (c - w0 - w1)];
  out[i] = v;
}

__global__ void vec_add(float* __restrict__ a, const float* __restrict__ b, int n) {
  int i = blockIdx.x * blockDim.x + threadIdx.x;
  if (i < n) a[i] += b[i];
}

// ===========================================================================
// Host orchestration
// ===========================================================================
static inline long cdiv(long a, long b) { return (a + b - 1) / b; }
static inline int ilog2(int v) { int s = 0; while ((1 << s) < v) ++s; return s; }

extern "C" void kernel_launch(void* const* d_in, const int* in_sizes, int n_in,
                              void* d_out, int out_size, void* d_ws, size_t ws_size,
                              hipStream_t stream) {
  (void)in_sizes; (void)n_in; (void)out_size; (void)ws_size;
  const int NV = 162, NF = 320, LAT = 32;

  const float* x_in  = (const float*)d_in[0];
  const float* sv_in = (const float*)d_in[2];
  const int*   faces = (const int*)d_in[3];

  int cur = 4;
  auto nx = [&]() { return (const float*)d_in[cur++]; };

  struct ConvP { const float *b1, *b2, *w1, *w2; };
  struct GC    { const float *bs, *wn, *ws; };
  struct SkipP { const float *sd_b, *sd_w, *sum_b, *sum_w; };
  struct UpP   { const float *b1, *b2, *upb, *upw, *w1, *w2; };

  ConvP down[5];
  for (int i = 0; i < 5; ++i) { down[i].b1 = nx(); down[i].b2 = nx(); down[i].w1 = nx(); down[i].w2 = nx(); }
  GC f2f[5][4];
  for (int i = 0; i < 5; ++i)
    for (int j = 0; j < 4; ++j) { f2f[i][j].bs = nx(); f2f[i][j].wn = nx(); f2f[i][j].ws = nx(); }
  GC f2v[5][3];
  for (int i = 0; i < 5; ++i)
    for (int j = 0; j < 3; ++j) { f2v[i][j].bs = nx(); f2v[i][j].wn = nx(); f2v[i][j].ws = nx(); }
  const float* final_b = nx();
  const float* final_w = nx();
  SkipP skipP[5];
  for (int i = 0; i < 5; ++i) { skipP[i].sd_b = nx(); skipP[i].sd_w = nx(); skipP[i].sum_b = nx(); skipP[i].sum_w = nx(); }
  UpP upP[4];
  for (int i = 0; i < 4; ++i) { upP[i].b1 = nx(); upP[i].b2 = nx(); upP[i].upb = nx(); upP[i].upw = nx(); upP[i].w1 = nx(); upP[i].w2 = nx(); }

  // ---- workspace bump allocator ----
  char* wsb = (char*)d_ws;
  size_t off = 0;
  auto allocB = [&](size_t bytes) { void* p = wsb + off; off = (off + bytes + 255) & ~(size_t)255; return p; };
  auto allocF = [&](size_t elems) { return (float*)allocB(elems * sizeof(float)); };
  auto allocH = [&](size_t elems) { return (__bf16*)allocB(elems * sizeof(__bf16)); };
  auto allocU = [&](size_t elems) { return (unsigned*)allocB(elems * sizeof(unsigned)); };

  float* bigA = allocF(8LL * 1024 * 1024 + 65536);
  float* bigB = allocF(4LL * 1024 * 1024 + 262144);
  float* bigD = allocF(4LL * 1024 * 1024 + 262144);
  float* hb0  = allocF(4LL * 1024 * 1024 + 262144);
  float* hb1  = allocF(4LL * 1024 * 1024 + 262144);
  unsigned* pairBuf = allocU(4LL * 1024 * 1024 + 262144);  // bf16-pair staging
  float* dn[5];
  for (int i = 0; i < 5; ++i) {
    long d = 64 >> i, C = 16 << i;
    dn[i] = allocF(C * d * d * d);
  }
  float* verts  = allocF(NV * 3);
  float* An     = allocF((long)NV * NV);
  float* center = allocF((long)NV * 256);
  float* neigh  = allocF((long)NV * 27 * 3);
  float* feats  = allocF((long)NV * 27 * 256);
  float* skipd  = allocF((long)NV * 256);
  float* latent = allocF((long)NV * LAT);
  float* zA     = allocF((long)NV * 300);
  float* zB     = allocF((long)NV * 300);
  float* nxb    = allocF((long)NV * 300);
  float* dv     = allocF((long)NV * 16);

  auto packSize = [&](long Co, long K) { return cdiv(Co, 16) * cdiv(K, 32) * 512; };
  __bf16 *pk_dn1[5], *pk_dn2[5], *pk_up1[4], *pk_up2[4];
  for (int i = 0; i < 5; ++i) {
    long C = 16 << i, Ci = (i == 0) ? 1 : (8 << i);
    pk_dn1[i] = allocH(packSize(C, Ci * 27));
    pk_dn2[i] = allocH(packSize(C, C * 27));
  }
  for (int i = 0; i < 4; ++i) {
    long ci = 16 << (4 - i), co = ci >> 1;
    pk_up1[i] = allocH(packSize(co, ci * 27));
    pk_up2[i] = allocH(packSize(co, co * 27));
  }

  auto packW = [&](const float* w, __bf16* dst, int Co, int Ci) {
    long total = packSize(Co, (long)Ci * 27);
    pack_wmma_a<<<dim3((unsigned)cdiv(total, 256)), dim3(256), 0, stream>>>(
        w, dst, Co, Ci, ilog2(Ci));
  };
  for (int i = 0; i < 5; ++i) {
    int C = 16 << i, Ci = (i == 0) ? 1 : (8 << i);
    packW(down[i].w1, pk_dn1[i], C, Ci);
    packW(down[i].w2, pk_dn2[i], C, C);
  }
  for (int i = 0; i < 4; ++i) {
    int ci = 16 << (4 - i), co = ci >> 1;
    packW(upP[i].w1, pk_up1[i], co, ci);
    packW(upP[i].w2, pk_up2[i], co, co);
  }

  // conv3 on an f32 input (Ci >= 16): pack pairs, then templated WMMA conv
  auto conv3 = [&](const float* in, const __bf16* apack, const float* bias,
                   float* out, int Ci, int Co, int d, int relu) {
    long DHW = (long)d * d * d;
    long ptot = (long)(Ci / 2) * DHW;
    pack_pairs<<<dim3((unsigned)cdiv(ptot, 256)), dim3(256), 0, stream>>>(
        in, pairBuf, Ci / 2, DHW);
    int tiles = Co / 16;
    dim3 blk(32);
    int cs = ilog2(Ci);
    if (tiles % 4 == 0)
      conv3_wmma_pair<4><<<dim3((unsigned)cdiv(DHW, 16), tiles / 4), blk, 0, stream>>>(
          pairBuf, apack, bias, out, Ci, cs, d, relu);
    else if (tiles == 2)
      conv3_wmma_pair<2><<<dim3((unsigned)cdiv(DHW, 16), 1), blk, 0, stream>>>(
          pairBuf, apack, bias, out, Ci, cs, d, relu);
    else
      conv3_wmma_pair<1><<<dim3((unsigned)cdiv(DHW, 16), 1), blk, 0, stream>>>(
          pairBuf, apack, bias, out, Ci, cs, d, relu);
  };
  auto gemm = [&](const float* X, const float* W, const float* bias, float* out,
                  int M, int K, int N, int acc, int relu) {
    long total = (long)M * N;
    gemm_small<<<dim3((unsigned)cdiv(total, 256)), dim3(256), 0, stream>>>(X, W, bias, out, M, K, N, acc, relu);
  };
  auto gconv = [&](const float* X, int K, const GC& p, int N, float* Y, int relu) {
    gemm(X, p.ws, p.bs, Y, NV, K, N, 0, 0);
    gemm(An, X, nullptr, nxb, NV, NV, K, 0, 0);
    gemm(nxb, p.wn, nullptr, Y, NV, K, N, 1, relu);
  };

  // ---- adjacency ----
  adj_zero<<<dim3((unsigned)cdiv((long)NV * NV, 256)), dim3(256), 0, stream>>>(An, NV);
  adj_edges<<<dim3((unsigned)cdiv(NF * 3, 256)), dim3(256), 0, stream>>>(faces, An, NF, NV);
  adj_norm<<<dim3(1), dim3(256), 0, stream>>>(An, NV);
  hipMemcpyAsync(verts, sv_in, NV * 3 * sizeof(float), hipMemcpyDeviceToDevice, stream);

  // ---- down path ----
  conv3_wmma_c1<<<dim3((unsigned)cdiv(64LL * 64 * 64, 16)), dim3(32), 0, stream>>>(
      x_in, pk_dn1[0], down[0].b1, bigB, 64, 1);
  conv3(bigB, pk_dn2[0], down[0].b2, dn[0], 16, 16, 64, 1);
  for (int i = 1; i < 5; ++i) {
    int dIn = 64 >> (i - 1), dOut = 64 >> i;
    int Cin = 16 << (i - 1), Cout = 16 << i;
    long tot = (long)Cin * dOut * dOut * dOut;
    maxpool2<<<dim3((unsigned)cdiv(tot, 256)), dim3(256), 0, stream>>>(dn[i - 1], bigB, Cin, dIn);
    conv3(bigB, pk_dn1[i], down[i].b1, bigD, Cin, Cout, dOut, 1);
    conv3(bigD, pk_dn2[i], down[i].b2, dn[i], Cout, Cout, dOut, 1);
  }

  // ---- up path interleaved with mesh head ----
  const float* h = dn[4];
  for (int i = 0; i <= 4; ++i) {
    if (i > 0) {
      const UpP& u = upP[i - 1];
      int ci = 16 << (5 - i), co = ci >> 1;
      int di = 4 << (i - 1), dop = di << 1;
      long outTot = (long)co * dop * dop * dop;
      upconv2<<<dim3((unsigned)cdiv(outTot, 256)), dim3(256), 0, stream>>>(h, u.upw, u.upb, bigB, ci, co, di);
      size_t blk = (size_t)co * dop * dop * dop * sizeof(float);
      hipMemcpyAsync(bigA, dn[4 - i], blk, hipMemcpyDeviceToDevice, stream);
      hipMemcpyAsync((char*)bigA + blk, bigB, blk, hipMemcpyDeviceToDevice, stream);
      conv3(bigA, pk_up1[i - 1], u.b1, bigD, 2 * co, co, dop, 1);
      float* hNew = ((i - 1) & 1) ? hb1 : hb0;
      conv3(bigD, pk_up2[i - 1], u.b2, hNew, co, co, dop, 1);
      h = hNew;
    }
    int d = 4 << i;
    int C = 256 >> i;
    trilinear<<<dim3((unsigned)cdiv((long)NV * C, 256)), dim3(256), 0, stream>>>(h, C, d, verts, NV, center);
    build_neigh<<<dim3((unsigned)cdiv(NV * 27, 128)), dim3(128), 0, stream>>>(
        center, skipP[i].sd_w, skipP[i].sd_b, verts, neigh, NV, C, d);
    trilinear<<<dim3((unsigned)cdiv((long)NV * 27 * C, 256)), dim3(256), 0, stream>>>(h, C, d, neigh, NV * 27, feats);
    neigh_sum<<<dim3((unsigned)cdiv((long)NV * C, 128)), dim3(128), 0, stream>>>(
        feats, skipP[i].sum_w, skipP[i].sum_b, skipd, NV, C);
    int ciG = C + 3 + (i ? LAT : 0);
    if (i == 0)
      concat_cols<<<dim3((unsigned)cdiv((long)NV * ciG, 256)), dim3(256), 0, stream>>>(
          skipd, C, verts, 3, nullptr, 0, zA, NV);
    else
      concat_cols<<<dim3((unsigned)cdiv((long)NV * ciG, 256)), dim3(256), 0, stream>>>(
          latent, LAT, skipd, C, verts, 3, zA, NV);
    gconv(zA, ciG, f2f[i][0], LAT, zB, 1);
    gconv(zB, LAT, f2f[i][1], LAT, zA, 1);
    gconv(zA, LAT, f2f[i][2], LAT, zB, 1);
    gconv(zB, LAT, f2f[i][3], LAT, latent, 0);
    gconv(latent, LAT, f2v[i][0], 16, zA, 1);
    gconv(zA, 16, f2v[i][1], 8, zB, 1);
    gconv(zB, 8, f2v[i][2], 3, dv, 0);
    vec_add<<<dim3(2), dim3(256), 0, stream>>>(verts, dv, NV * 3);
  }

  float* outF = (float*)d_out;
  conv1x1<<<dim3((unsigned)cdiv(2LL * 64 * 64 * 64, 256)), dim3(256), 0, stream>>>(
      h, final_w, final_b, outF + NV * 3, 16, 2, 64LL * 64 * 64);
  hipMemcpyAsync(outF, verts, NV * 3 * sizeof(float), hipMemcpyDeviceToDevice, stream);
}
